// MultiScaleMaskedTransformerDecoder_34213709480562
// MI455X (gfx1250) — matrix operand order repaired and checked by reading, not compiled, not run on hardware
//
#include <hip/hip_runtime.h>

// ---------------------------------------------------------------------------
// MI455X (gfx1250) implementation.
// Big GEMM (131072x256x256) and the 8-head similarity einsum (K=32) run on
// v_wmma_f32_16x16x32_f16 (fp32 accumulate). The GEMM stages A/B K-slices
// into LDS with double-buffered GLOBAL_LOAD_ASYNC_TO_LDS_B128 (ASYNCcnt /
// s_wait_asynccnt), the CDNA5 async data-movement path. Argmax over N=16384
// uses packed (ordered-float<<32 | ~index) u64 keys reduced with ds_max_u64
// in LDS and global_atomic_max_u64 across workgroups (tie-break = smallest
// index, like jnp.argmax). Everything else is bandwidth-trivial VALU/LDS.
// ---------------------------------------------------------------------------

typedef __attribute__((ext_vector_type(16))) _Float16 v16h;
typedef __attribute__((ext_vector_type(8)))  _Float16 v8h;
typedef __attribute__((ext_vector_type(8)))  float    v8f;

static constexpr int Bb    = 8;
static constexpr int Cc    = 256;
static constexpr int Hh    = 128;
static constexpr int Ww    = 128;
static constexpr int Qq    = 100;
static constexpr int Nn    = Hh * Ww;        // 16384
static constexpr int HEADS = 8;
static constexpr int Dd    = 32;
static constexpr int QPAD  = 112;            // 7 tiles of 16

#if __has_builtin(__builtin_amdgcn_global_load_async_to_lds_b128)
#define HAVE_ASYNC_LDS 1
#else
#define HAVE_ASYNC_LDS 0
#endif

typedef int v4i __attribute__((__vector_size__(4 * sizeof(int))));
typedef __attribute__((address_space(1))) v4i gv4i;   // global
typedef __attribute__((address_space(3))) v4i lv4i;   // LDS

__device__ __forceinline__ void cp16_async(const _Float16* g, _Float16* l) {
#if HAVE_ASYNC_LDS
  // generic->AS1 keeps the 64-bit value; generic->AS3 is the low 32 bits
  // (ISA 10.2: LDS aperture address = {SHARED_BASE, offset[31:0]}).
  __builtin_amdgcn_global_load_async_to_lds_b128(
      (gv4i*)(unsigned long long)(uintptr_t)g,
      (lv4i*)(unsigned int)(uintptr_t)l, 0, 0);
#else
  *(v8h*)l = *(const v8h*)g;
#endif
}

template <int N>
__device__ __forceinline__ void wait_async() {
#if HAVE_ASYNC_LDS
#if __has_builtin(__builtin_amdgcn_s_wait_asynccnt)
  __builtin_amdgcn_s_wait_asynccnt(N);
#else
  asm volatile("s_wait_asynccnt %0" ::"n"(N) : "memory");
#endif
#endif
}

__device__ __forceinline__ unsigned long long packKey(float s, unsigned n) {
  unsigned u = __float_as_uint(s);
  u = (u & 0x80000000u) ? ~u : (u | 0x80000000u);   // order-preserving float->uint
  return ((unsigned long long)u << 32) | (unsigned long long)(0xFFFFFFFFu - n);
}

// ---------------- init: weights->f16, zero argmax keys, flags=1 -------------
__global__ __launch_bounds__(256) void k_init(const float* __restrict__ lrw,
                                              _Float16* __restrict__ lrw16,
                                              unsigned long long* __restrict__ keys,
                                              int* __restrict__ flags) {
  int tid = blockIdx.x * 256 + threadIdx.x;
  if (tid < Cc * Cc) lrw16[tid] = (_Float16)lrw[tid];
  if (tid < Bb * HEADS * QPAD) keys[tid] = 0ull;
  if (tid < Bb) flags[tid] = 1;
}

// ---------------- BatchNorm stats over (B,H,W) per channel ------------------
__global__ __launch_bounds__(256) void k_bnstats(const float* __restrict__ mem,
                                                 const float* __restrict__ pos,
                                                 const float* __restrict__ bn_g,
                                                 const float* __restrict__ bn_b,
                                                 float* __restrict__ scale,
                                                 float* __restrict__ shift) {
  __shared__ float r1[256], r2[256];
  int c = blockIdx.x, tid = threadIdx.x;
  float s1 = 0.f, s2 = 0.f;
  for (int i = tid; i < Bb * Nn; i += 256) {
    int b = i >> 14, hw = i & (Nn - 1);
    size_t idx = ((size_t)(b * Cc + c) << 14) + hw;
    float v = mem[idx] + pos[idx];
    s1 += v; s2 += v * v;
  }
  r1[tid] = s1; r2[tid] = s2; __syncthreads();
  for (int s = 128; s > 0; s >>= 1) {
    if (tid < s) { r1[tid] += r1[tid + s]; r2[tid] += r2[tid + s]; }
    __syncthreads();
  }
  if (tid == 0) {
    float cnt = (float)(Bb * Nn);
    float mu  = r1[0] / cnt;
    float var = r2[0] / cnt - mu * mu;
    float r   = rsqrtf(var + 1e-5f);
    float sc  = r * bn_g[c];
    scale[c] = sc;
    shift[c] = bn_b[c] - mu * sc;
  }
}

// ---------------- depthwise 3x3 conv (zero pad) on normalized input ---------
__global__ __launch_bounds__(256) void k_conv(const float* __restrict__ mem,
                                              const float* __restrict__ pos,
                                              const float* __restrict__ scale,
                                              const float* __restrict__ shift,
                                              const float* __restrict__ conv_w,
                                              const float* __restrict__ conv_b,
                                              float* __restrict__ out) {
  __shared__ float img[Nn];                    // 64 KB of the 320 KB WGP LDS
  int bc = blockIdx.x;
  int b = bc >> 8, c = bc & 255;
  int tid = threadIdx.x;
  float sc = scale[c], sh = shift[c];
  size_t base = ((size_t)(b * Cc + c)) << 14;
  for (int i = tid; i < Nn; i += 256)
    img[i] = (mem[base + i] + pos[base + i]) * sc + sh;
  float w9[9];
#pragma unroll
  for (int k = 0; k < 9; ++k) w9[k] = conv_w[c * 9 + k];
  float bias = conv_b[c];
  __syncthreads();
  for (int i = tid; i < Nn; i += 256) {
    int h = i >> 7, w = i & 127;
    float acc = bias;
#pragma unroll
    for (int dy = -1; dy <= 1; ++dy) {
#pragma unroll
      for (int dx = -1; dx <= 1; ++dx) {
        int hh = h + dy, ww = w + dx;
        if (hh >= 0 && hh < Hh && ww >= 0 && ww < Ww)
          acc += img[(hh << 7) + ww] * w9[(dy + 1) * 3 + (dx + 1)];
      }
    }
    out[base + i] = acc;
  }
}

// ---------------- (B,C,N) fp32 -> (B,N,C) f16 tile transpose ----------------
__global__ __launch_bounds__(256) void k_transpose(const float* __restrict__ src,
                                                   _Float16* __restrict__ dst) {
  __shared__ float sm[32][33];
  int n0 = blockIdx.x * 32, c0 = blockIdx.y * 32, b = blockIdx.z;
  int lr = threadIdx.x >> 5, lc = threadIdx.x & 31;
#pragma unroll
  for (int rr = 0; rr < 4; ++rr) {
    int cl = rr * 8 + lr;
    sm[cl][lc] = src[((size_t)(b * Cc + c0 + cl) << 14) + n0 + lc];
  }
  __syncthreads();
#pragma unroll
  for (int rr = 0; rr < 4; ++rr) {
    int nl = rr * 8 + lr;
    dst[((size_t)(b * Nn + n0 + nl) << 8) + c0 + lc] = (_Float16)sm[lc][nl];
  }
}

// ---------------- big GEMM: (B*N,256) x (256,256)^T, f16 WMMA ---------------
// Double-buffered async global->LDS staging of the 128x32 A and B K-slices.
__global__ __launch_bounds__(256) void k_gemm(const _Float16* __restrict__ A16,
                                              const _Float16* __restrict__ W16,
                                              const float* __restrict__ bias,
                                              float* __restrict__ out) {
  __shared__ __align__(64) _Float16 As[2][128][32];   // 16 KB
  __shared__ __align__(64) _Float16 Bs[2][128][32];   // 16 KB
  int tid = threadIdx.x, wv = tid >> 5, lane = tid & 31, l16 = lane & 15;
  int wm = wv & 3, wn = wv >> 2;
  size_t mBlock = (size_t)blockIdx.x * 128;
  int nBlock = blockIdx.y * 128;

  auto issue = [&](int ks, int buf) {
    int kb = ks * 32;
#pragma unroll
    for (int r = 0; r < 2; ++r) {
      int idx = tid + r * 256;            // 512 16-byte chunks per tile
      int row = idx >> 2, part = (idx & 3) * 8;
      cp16_async(A16 + ((mBlock + row) << 8) + kb + part, &As[buf][row][part]);
      cp16_async(W16 + ((size_t)(nBlock + row) << 8) + kb + part, &Bs[buf][row][part]);
    }
  };

  v8f acc[2][4];
#pragma unroll
  for (int mt = 0; mt < 2; ++mt)
#pragma unroll
    for (int nt = 0; nt < 4; ++nt)
#pragma unroll
      for (int i = 0; i < 8; ++i) acc[mt][nt][i] = 0.f;

  int kA0 = (lane < 16) ? 0 : 16;     // A 16x32 f16 fragment: contiguous k = 16h + s
  int kB0 = (lane < 16) ? 0 : 16;     // B 32x16 f16 fragment layout
  issue(0, 0);
  for (int ks = 0; ks < 8; ++ks) {
    int buf = ks & 1;
    if (ks < 7) {                     // 4 newer ops in flight, current 4 landed
      issue(ks + 1, buf ^ 1);
      wait_async<4>();                // in-order async completion (ISA 08 §4.1)
    } else {
      wait_async<0>();
    }
    __syncthreads();                  // all waves' LDS writes visible

    v16h a[2], bb[4];
#pragma unroll
    for (int mt = 0; mt < 2; ++mt) {
      const _Float16* ap = &As[buf][wm * 32 + mt * 16 + l16][0];
      v8h lo = *(const v8h*)(ap + kA0);
      v8h hi = *(const v8h*)(ap + kA0 + 8);
#pragma unroll
      for (int i = 0; i < 8; ++i) { a[mt][i] = lo[i]; a[mt][8 + i] = hi[i]; }
    }
#pragma unroll
    for (int nt = 0; nt < 4; ++nt)
      bb[nt] = *(const v16h*)(&Bs[buf][wn * 64 + nt * 16 + l16][kB0]);
#pragma unroll
    for (int mt = 0; mt < 2; ++mt)
#pragma unroll
      for (int nt = 0; nt < 4; ++nt)
        acc[mt][nt] = __builtin_amdgcn_wmma_f32_16x16x32_f16(
            false, a[mt], false, bb[nt], (short)0, acc[mt][nt], false, false);
    __syncthreads();                  // done reading before buf is overwritten
  }
  int moff = (lane >= 16) ? 8 : 0;
#pragma unroll
  for (int mt = 0; mt < 2; ++mt)
#pragma unroll
    for (int nt = 0; nt < 4; ++nt) {
      int co = nBlock + wn * 64 + nt * 16 + l16;
      float bv = bias[co];
#pragma unroll
      for (int v = 0; v < 8; ++v)
        out[((mBlock + wm * 32 + mt * 16 + moff + v) << 8) + co] = acc[mt][nt][v] + bv;
    }
}

// ---------------- row L2-norm: fp32 in place + f16 copy ---------------------
__global__ __launch_bounds__(256) void k_rownorm(float* __restrict__ buf,
                                                 _Float16* __restrict__ out16) {
  __shared__ float red[256];
  size_t m = blockIdx.x;
  int tid = threadIdx.x;
  float v = buf[(m << 8) + tid];
  red[tid] = v * v; __syncthreads();
  for (int s = 128; s > 0; s >>= 1) {
    if (tid < s) red[tid] += red[tid + s];
    __syncthreads();
  }
  float sc = 1.f / fmaxf(sqrtf(red[0]), 1e-12f);
  float nv = v * sc;
  buf[(m << 8) + tid] = nv;
  out16[(m << 8) + tid] = (_Float16)nv;
}

// ---------------- query path: LN -> linear -> l2norm ------------------------
__global__ __launch_bounds__(256) void k_query(const float* __restrict__ tgt,
                                               const float* __restrict__ qpos,
                                               const float* __restrict__ g,
                                               const float* __restrict__ be,
                                               const float* __restrict__ qw,
                                               const float* __restrict__ qb,
                                               float* __restrict__ tqf,
                                               _Float16* __restrict__ tq16) {
  __shared__ float y[256], red[256];
  int b = blockIdx.x / Qq, q = blockIdx.x % Qq, c = threadIdx.x;
  size_t src = ((size_t)(q * Bb + b) << 8) + c;
  float x = tgt[src] + qpos[src];
  red[c] = x; __syncthreads();
  for (int s = 128; s > 0; s >>= 1) { if (c < s) red[c] += red[c + s]; __syncthreads(); }
  float mu = red[0] / 256.f; __syncthreads();
  red[c] = (x - mu) * (x - mu); __syncthreads();
  for (int s = 128; s > 0; s >>= 1) { if (c < s) red[c] += red[c + s]; __syncthreads(); }
  float var = red[0] / 256.f; __syncthreads();
  y[c] = (x - mu) * rsqrtf(var + 1e-5f) * g[c] + be[c];
  __syncthreads();
  float o = qb[c];
  for (int k = 0; k < 256; ++k) o += y[k] * qw[(c << 8) + k];
  red[c] = o * o; __syncthreads();
  for (int s = 128; s > 0; s >>= 1) { if (c < s) red[c] += red[c + s]; __syncthreads(); }
  float sc = 1.f / fmaxf(sqrtf(red[0]), 1e-12f);
  float nv = o * sc;
  size_t dst = ((size_t)(b * Qq + q) << 8) + c;
  tqf[dst] = nv;
  tq16[dst] = (_Float16)nv;
}

// ---------------- all-masked flag per batch ---------------------------------
__global__ __launch_bounds__(256) void k_maskflags(const float* __restrict__ mask,
                                                   int* __restrict__ flags) {
  __shared__ float red[256];
  int b = blockIdx.x / Qq, q = blockIdx.x % Qq, tid = threadIdx.x;
  float mx = -3.4e38f;
  size_t base = (size_t)(b * Qq + q) << 14;
  for (int i = tid; i < Nn; i += 256) mx = fmaxf(mx, mask[base + i]);
  red[tid] = mx; __syncthreads();
  for (int s = 128; s > 0; s >>= 1) { if (tid < s) red[tid] = fmaxf(red[tid], red[tid + s]); __syncthreads(); }
  if (tid == 0 && red[0] >= 0.f) atomicAnd(&flags[b], 0);  // this (b,q) not fully masked
}

// ---------------- similarity (WMMA K=32) + masked argmax --------------------
__global__ __launch_bounds__(256) void k_sim(const _Float16* __restrict__ X16,
                                             const _Float16* __restrict__ Q16,
                                             const float* __restrict__ mask,
                                             const int* __restrict__ flags,
                                             unsigned long long* __restrict__ gkeys) {
  __shared__ unsigned long long bestQ[QPAD];
  int bh = blockIdx.x, b = bh >> 3, h = bh & 7;
  int tid = threadIdx.x, wv = tid >> 5, lane = tid & 31, l16 = lane & 15;
  if (tid < QPAD) bestQ[tid] = 0ull;
  __syncthreads();
  bool skipMask = flags[b] != 0;
  int nbaseWave = blockIdx.y * 1024 + wv * 128;
  int kA0 = (lane < 16) ? 0 : 16;
  int moff = (lane < 16) ? 0 : 8;
  for (int qt = 0; qt < QPAD / 16; ++qt) {
    int qcol = qt * 16 + l16;
    v16h bq;
    if (qcol < Qq) {
      bq = *(const v16h*)(Q16 + ((size_t)(b * Qq + qcol) << 8) + h * Dd + (lane < 16 ? 0 : 16));
    } else {
#pragma unroll
      for (int i = 0; i < 16; ++i) bq[i] = (_Float16)0.f;
    }
    unsigned long long bestKey = 0ull;
    for (int it = 0; it < 8; ++it) {
      int rowBase = nbaseWave + it * 16;
      const _Float16* ap = X16 + ((size_t)(b * Nn + rowBase + l16) << 8) + h * Dd;
      __builtin_prefetch(ap + (16 << 8), 0, 1);
      v8h lo = *(const v8h*)(ap + kA0);
      v8h hi = *(const v8h*)(ap + kA0 + 8);
      v16h a;
#pragma unroll
      for (int i = 0; i < 8; ++i) { a[i] = lo[i]; a[8 + i] = hi[i]; }
      v8f cz;
#pragma unroll
      for (int i = 0; i < 8; ++i) cz[i] = 0.f;
      v8f c = __builtin_amdgcn_wmma_f32_16x16x32_f16(false, a, false, bq,
                                                     (short)0, cz, false, false);
      if (qcol < Qq) {
        const float* mp = mask + ((size_t)(b * Qq + qcol) << 14) + rowBase + moff;
        float4 m0 = *(const float4*)mp;
        float4 m1 = *(const float4*)(mp + 4);
        float mv[8] = {m0.x, m0.y, m0.z, m0.w, m1.x, m1.y, m1.z, m1.w};
#pragma unroll
        for (int v = 0; v < 8; ++v) {
          if (skipMask || mv[v] >= 0.f) {
            unsigned long long key = packKey(c[v], (unsigned)(rowBase + moff + v));
            if (key > bestKey) bestKey = key;
          }
        }
      }
    }
    if (qcol < Qq && bestKey) atomicMax(&bestQ[qt * 16 + l16], bestKey);  // ds_max_u64
  }
  __syncthreads();
  for (int t = tid; t < Qq; t += 256) {
    unsigned long long v = bestQ[t];
    if (v) atomicMax(&gkeys[(size_t)bh * QPAD + t], v);  // global_atomic_max_u64
  }
}

// ---------------- gather sel, y1 = (sel*tq) @ proj_w.T + proj_b -------------
__global__ __launch_bounds__(256) void k_selproj(const unsigned long long* __restrict__ gkeys,
                                                 const float* __restrict__ memn,
                                                 const float* __restrict__ tqf,
                                                 const float* __restrict__ pw,
                                                 const float* __restrict__ pb,
                                                 float* __restrict__ selb,
                                                 float* __restrict__ y1) {
  __shared__ float prod[256];
  int b = blockIdx.x / Qq, q = blockIdx.x % Qq, c = threadIdx.x;
  int h = c >> 5;
  unsigned long long key = gkeys[(size_t)(b * HEADS + h) * QPAD + q];
  unsigned n = key ? (0xFFFFFFFFu - (unsigned)(key & 0xFFFFFFFFull)) : 0u;
  float sv = memn[((size_t)b * Nn + n) * 256 + c];
  float tv = tqf[((size_t)(b * Qq + q) << 8) + c];
  size_t dst = ((size_t)(b * Qq + q) << 8) + c;
  selb[dst] = sv;
  prod[c] = sv * tv;
  __syncthreads();
  float o = pb[c];
  for (int k = 0; k < 256; ++k) o += prod[k] * pw[(c << 8) + k];
  y1[dst] = o;
}

// ---------------- L2 norm across Q (axis=1) per (b,c) -----------------------
__global__ __launch_bounds__(256) void k_colnorm(const float* __restrict__ y1,
                                                 float* __restrict__ cinv) {
  int b = blockIdx.x, c = threadIdx.x;
  float s = 0.f;
  for (int q = 0; q < Qq; ++q) {
    float v = y1[((size_t)(b * Qq + q) << 8) + c];
    s += v * v;
  }
  cinv[(b << 8) + c] = 1.f / fmaxf(sqrtf(s), 1e-12f);
}

// ---------------- final: *alpha + sel, @ final_w.T + b, + residual ----------
__global__ __launch_bounds__(256) void k_final(const float* __restrict__ y1,
                                               const float* __restrict__ cinv,
                                               const float* __restrict__ alpha,
                                               const float* __restrict__ selb,
                                               const float* __restrict__ fw,
                                               const float* __restrict__ fb,
                                               const float* __restrict__ tgt,
                                               float* __restrict__ out) {
  __shared__ float u[256];
  int b = blockIdx.x / Qq, q = blockIdx.x % Qq, c = threadIdx.x;
  size_t src = ((size_t)(b * Qq + q) << 8) + c;
  u[c] = y1[src] * cinv[(b << 8) + c] * alpha[c] + selb[src];
  __syncthreads();
  float o = fb[c];
  for (int k = 0; k < 256; ++k) o += u[k] * fw[(c << 8) + k];
  size_t dst = ((size_t)(q * Bb + b) << 8) + c;
  out[dst] = o + tgt[dst];
}

// ---------------------------------------------------------------------------
extern "C" void kernel_launch(void* const* d_in, const int* in_sizes, int n_in,
                              void* d_out, int out_size, void* d_ws, size_t ws_size,
                              hipStream_t stream) {
  const float* tgt     = (const float*)d_in[0];
  const float* memory  = (const float*)d_in[1];
  const float* mmask   = (const float*)d_in[2];
  const float* pos     = (const float*)d_in[3];
  const float* qpos    = (const float*)d_in[4];
  const float* conv_w  = (const float*)d_in[5];
  const float* conv_b  = (const float*)d_in[6];
  const float* bn_g    = (const float*)d_in[7];
  const float* bn_b    = (const float*)d_in[8];
  const float* lr_w    = (const float*)d_in[9];
  const float* lr_b    = (const float*)d_in[10];
  const float* qln_g   = (const float*)d_in[11];
  const float* qln_b   = (const float*)d_in[12];
  const float* qp_w    = (const float*)d_in[13];
  const float* qp_b    = (const float*)d_in[14];
  const float* proj_w  = (const float*)d_in[15];
  const float* proj_b  = (const float*)d_in[16];
  const float* final_w = (const float*)d_in[17];
  const float* final_b = (const float*)d_in[18];
  const float* alpha   = (const float*)d_in[19];
  float* outp = (float*)d_out;

  // workspace layout (all 256B aligned)
  constexpr size_t SZ1   = (size_t)Bb * Nn * Cc * 4;   // 128 MB fp32
  constexpr size_t SZ2   = (size_t)Bb * Nn * Cc * 2;   //  64 MB f16
  constexpr size_t O_B1  = 0;
  constexpr size_t O_B2  = O_B1 + SZ1;
  constexpr size_t O_LW  = O_B2 + SZ2;                 // lr_w f16
  constexpr size_t O_SC  = O_LW + (size_t)Cc * Cc * 2;
  constexpr size_t O_SH  = O_SC + 1024;
  constexpr size_t O_TQ  = O_SH + 1024;                // tq fp32
  constexpr size_t O_T16 = O_TQ + (size_t)Bb * Qq * Cc * 4;
  constexpr size_t O_KEY = O_T16 + ((size_t)Bb * Qq * Cc * 2 + 255) / 256 * 256;
  constexpr size_t O_FLG = O_KEY + (size_t)Bb * HEADS * QPAD * 8;
  constexpr size_t O_SEL = O_FLG + 256;
  constexpr size_t O_Y1  = O_SEL + (size_t)Bb * Qq * Cc * 4;
  constexpr size_t O_CI  = O_Y1 + (size_t)Bb * Qq * Cc * 4;
  constexpr size_t WS_REQ = O_CI + (size_t)Bb * Cc * 4;
  if (ws_size < WS_REQ) return;

  char* ws = (char*)d_ws;
  float*              buf1  = (float*)(ws + O_B1);
  _Float16*           buf2  = (_Float16*)(ws + O_B2);
  _Float16*           lrw16 = (_Float16*)(ws + O_LW);
  float*              scale = (float*)(ws + O_SC);
  float*              shift = (float*)(ws + O_SH);
  float*              tqf   = (float*)(ws + O_TQ);
  _Float16*           tq16  = (_Float16*)(ws + O_T16);
  unsigned long long* keys  = (unsigned long long*)(ws + O_KEY);
  int*                flags = (int*)(ws + O_FLG);
  float*              selb  = (float*)(ws + O_SEL);
  float*              y1    = (float*)(ws + O_Y1);
  float*              cinv  = (float*)(ws + O_CI);

  k_init<<<dim3(256), 256, 0, stream>>>(lr_w, lrw16, keys, flags);
  k_bnstats<<<dim3(Cc), 256, 0, stream>>>(memory, pos, bn_g, bn_b, scale, shift);
  k_conv<<<dim3(Bb * Cc), 256, 0, stream>>>(memory, pos, scale, shift, conv_w, conv_b, buf1);
  k_transpose<<<dim3(Nn / 32, Cc / 32, Bb), 256, 0, stream>>>(buf1, buf2);
  k_gemm<<<dim3(Bb * Nn / 128, Cc / 128), 256, 0, stream>>>(buf2, lrw16, lr_b, buf1);
  k_rownorm<<<dim3(Bb * Nn), 256, 0, stream>>>(buf1, buf2);
  k_query<<<dim3(Bb * Qq), 256, 0, stream>>>(tgt, qpos, qln_g, qln_b, qp_w, qp_b, tqf, tq16);
  k_maskflags<<<dim3(Bb * Qq), 256, 0, stream>>>(mmask, flags);
  k_sim<<<dim3(Bb * HEADS, 16), 256, 0, stream>>>(buf2, tq16, mmask, flags, keys);
  k_selproj<<<dim3(Bb * Qq), 256, 0, stream>>>(keys, buf1, tqf, proj_w, proj_b, selb, y1);
  k_colnorm<<<dim3(Bb), 256, 0, stream>>>(y1, cinv);
  k_final<<<dim3(Bb * Qq), 256, 0, stream>>>(y1, cinv, alpha, selb, final_w, final_b, tgt, outp);
}
